// RandomBinaryConvV1_87557203296394
// MI455X (gfx1250) — matrix operand order repaired and checked by reading, not compiled, not run on hardware
//
#include <hip/hip_runtime.h>
#include <hip/hip_bf16.h>
#include <stdint.h>

#if defined(__has_builtin)
#if __has_builtin(__builtin_amdgcn_global_load_async_to_lds_b128)
#define HAVE_ASYNC 1
#endif
#endif
#ifndef HAVE_ASYNC
#define HAVE_ASYNC 0
#endif

typedef _Float16 v16h __attribute__((ext_vector_type(16)));
typedef _Float16 v8h  __attribute__((ext_vector_type(8)));
typedef float    v8f  __attribute__((ext_vector_type(8)));
typedef int      i4v  __attribute__((vector_size(16)));   // matches builtin param type

// Problem constants
#define BATCH 32
#define CIN   256
#define HH    56
#define WW    56
#define OUTC  512
#define HWSZ  (HH * WW)            // 3136
#define MTOT  (BATCH * HWSZ)       // 100352
#define KTAP  9
// Main-kernel tiling
#define MT    128
#define NT    128
#define ALDS_STRIDE 40             // 32 halves + 8 pad (80B rows spread banks)
#define A_BUF_H (MT * ALDS_STRIDE) // 5120 halves / buffer
#define B_BUF_H 4096               // 8 frags * 512 halves / buffer
#define NITER  (KTAP * 8)          // 72 k-steps

#if HAVE_ASYNC
__device__ __forceinline__ void async_b128(const _Float16* g, _Float16* l) {
  __builtin_amdgcn_global_load_async_to_lds_b128(
      (__attribute__((address_space(1))) i4v*)g,
      (__attribute__((address_space(3))) i4v*)l, 0, 0);
}
__device__ __forceinline__ void wait_async0() {
#if __has_builtin(__builtin_amdgcn_s_wait_asynccnt)
  __builtin_amdgcn_s_wait_asynccnt((short)0);
#else
  asm volatile("s_wait_asynccnt 0" ::: "memory");
#endif
}
#endif

// -------------------------------------------------------------------------
// Pre-pass 1: x NCHW f32 -> NHWC f16 (LDS-tiled transpose)
// grid (49, 8, 32), block 256
// -------------------------------------------------------------------------
__global__ __launch_bounds__(256) void k_nchw_to_nhwc_f16(
    const float* __restrict__ x, _Float16* __restrict__ xh) {
  __shared__ _Float16 tile[32][65];
  const int b   = blockIdx.z;
  const int c0  = blockIdx.y << 5;
  const int hw0 = blockIdx.x << 6;
  const int t   = threadIdx.x;
  {
    const int cc   = t >> 3;
    const int part = (t & 7) << 3;
    const float* src = x + ((size_t)(b * CIN + c0 + cc) * HWSZ + hw0 + part);
    float4 f0 = ((const float4*)src)[0];
    float4 f1 = ((const float4*)src)[1];
    tile[cc][part + 0] = (_Float16)f0.x;
    tile[cc][part + 1] = (_Float16)f0.y;
    tile[cc][part + 2] = (_Float16)f0.z;
    tile[cc][part + 3] = (_Float16)f0.w;
    tile[cc][part + 4] = (_Float16)f1.x;
    tile[cc][part + 5] = (_Float16)f1.y;
    tile[cc][part + 6] = (_Float16)f1.z;
    tile[cc][part + 7] = (_Float16)f1.w;
  }
  __syncthreads();
  {
    const int hwL = t & 63;
    const int cp  = (t >> 6) << 3;
    v8h v;
#pragma unroll
    for (int j = 0; j < 8; ++j) v[j] = tile[cp + j][hwL];
    *(v8h*)(xh + (size_t)(b * HWSZ + hw0 + hwL) * CIN + c0 + cp) = v;
  }
}

// -------------------------------------------------------------------------
// Pre-pass 2: ternary masks -> f16 B fragments, pre-swizzled to the 16-bit
// 32x16 B lane layout. wf[(((r*32 + n16)*8 + ks)*32 + lane)*16 + j]
// -------------------------------------------------------------------------
__global__ __launch_bounds__(256) void k_pack_weight(
    const unsigned char* __restrict__ pos, const unsigned char* __restrict__ neg,
    _Float16* __restrict__ wf) {
  const int t = blockIdx.x * 256 + threadIdx.x;       // 0 .. 73727
  const int lane = t & 31;
  const int ks   = (t >> 5) & 7;
  const int n16  = (t >> 8) & 31;
  const int r    = t >> 13;                           // 0..8
  const int n     = (n16 << 4) + (lane & 15);
  const int cbase = (ks << 5) + ((lane >> 4) << 4);
  v16h v;
#pragma unroll
  for (int j = 0; j < 16; ++j) {
    const int c  = cbase + j;
    const int kk = c * KTAP + r;
    const int p  = (int)pos[kk * OUTC + n];
    const int q  = (int)neg[kk * OUTC + n];
    v[j] = (_Float16)(p - q);
  }
  *(v16h*)(wf + ((size_t)(((r * 32 + n16) * 8 + ks) * 32 + lane) << 4)) = v;
}

// -------------------------------------------------------------------------
// Main implicit-GEMM conv. 128x128 block tile, 8 waves (4M x 2N), wave tile
// 32x64, V_WMMA_F32_16X16X32_F16. A and B double-buffered in LDS with a
// one-barrier software pipeline; async-to-LDS when the toolchain has it.
// grid (784, 4), block 256
// -------------------------------------------------------------------------
__global__ __launch_bounds__(256) void k_conv_wmma(
    const _Float16* __restrict__ xh, const _Float16* __restrict__ wf,
    float* __restrict__ out) {
  __shared__ __align__(16) _Float16 Asm[2 * A_BUF_H];
  __shared__ __align__(16) _Float16 Bsm[2 * B_BUF_H];

  const int tid   = threadIdx.x;
  const int lane  = tid & 31;
  const int wave  = tid >> 5;
  const int waveM = wave >> 1;
  const int waveN = wave & 1;
  const int mBase = blockIdx.x * MT;
  const int nBase = blockIdx.y * NT + waveN * 64;
  const int n16blk = blockIdx.y * 8;          // 8 B-fragments per block

  // A-loader coords: 2 threads per m-row, 16 channels each
  const int li = tid >> 1;
  const int cg = (tid & 1) << 4;
  const int m  = mBase + li;
  const int bb = m / HWSZ;
  const int hwm = m - bb * HWSZ;
  const int hh = hwm / WW;
  const int wwm = hwm - hh * WW;

  const int lrow = lane & 15;
  const int lhi  = lane >> 4;

  v8f acc[2][4];
#pragma unroll
  for (int f = 0; f < 2; ++f)
#pragma unroll
    for (int t2 = 0; t2 < 4; ++t2) acc[f][t2] = (v8f)0.0f;

  uint4 pA0 = make_uint4(0u, 0u, 0u, 0u);
  uint4 pA1 = make_uint4(0u, 0u, 0u, 0u);
  v16h  pB;

  // Fetch tile i (i = r*8 + ks) toward buffer i&1.
  auto fetchTile = [&](int i) {
    const int rr  = i >> 3;
    const int kss = i & 7;
    const int kh = (rr * 11) >> 5;            // rr/3 for rr in [0,8]
    const int kw = rr - kh * 3;
    const int hp = hh + kh - 1;
    const int wp = wwm + kw - 1;
    const bool v = ((unsigned)hp < (unsigned)HH) && ((unsigned)wp < (unsigned)WW);
    const int spix = m + (kh - 1) * WW + (kw - 1);          // shifted pixel index
    const _Float16* bs = wf +
        ((size_t)((rr * 32 + n16blk + (tid >> 5)) * 8 + kss) << 9) + ((tid & 31) << 4);
    const int ab = (i & 1) * A_BUF_H + li * ALDS_STRIDE + cg;
#if HAVE_ASYNC
    const int bbuf = (i & 1) * B_BUF_H + (tid << 4);
    async_b128(bs, &Bsm[bbuf]);
    async_b128(bs + 8, &Bsm[bbuf + 8]);
    if (v) {
      const _Float16* as = xh + ((size_t)spix * CIN + cg) + (kss << 5);
      async_b128(as, &Asm[ab]);
      async_b128(as + 8, &Asm[ab + 8]);
    } else {
      uint4 z = make_uint4(0u, 0u, 0u, 0u);
      ((uint4*)&Asm[ab])[0] = z;
      ((uint4*)&Asm[ab])[1] = z;
    }
#else
    pB = *(const v16h*)bs;
    if (v) {
      const _Float16* as = xh + ((size_t)spix * CIN + cg) + (kss << 5);
      pA0 = ((const uint4*)as)[0];
      pA1 = ((const uint4*)as)[1];
    } else {
      pA0 = make_uint4(0u, 0u, 0u, 0u);
      pA1 = make_uint4(0u, 0u, 0u, 0u);
    }
#endif
  };
  auto commitTile = [&](int i) {
#if !HAVE_ASYNC
    const int ab = (i & 1) * A_BUF_H + li * ALDS_STRIDE + cg;
    uint4* d = (uint4*)&Asm[ab];
    d[0] = pA0;
    d[1] = pA1;
    *(v16h*)&Bsm[(i & 1) * B_BUF_H + (tid << 4)] = pB;
#endif
  };

  // Prologue: tile 0 resident before the loop.
  fetchTile(0);
  commitTile(0);
#if HAVE_ASYNC
  wait_async0();
#endif
  __syncthreads();

#pragma unroll 1
  for (int it = 0; it < NITER; ++it) {
    // Prefetch next tile into the other buffer (overlaps with WMMA below).
    if (it < NITER - 1) fetchTile(it + 1);

    const int cb = it & 1;

    // B fragments from LDS (packed fragment layout, lane-contiguous 32B)
    const int bfb = cb * B_BUF_H + waveN * (4 * 512) + (lane << 4);
    v16h bf[4];
#pragma unroll
    for (int t2 = 0; t2 < 4; ++t2)
      bf[t2] = *(const v16h*)&Bsm[bfb + t2 * 512];

    // A fragments from LDS (16-bit 16x32 A lane layout)
    v16h af[2];
#pragma unroll
    for (int f = 0; f < 2; ++f) {
      const int row  = waveM * 32 + f * 16 + lrow;
      const int base = cb * A_BUF_H + row * ALDS_STRIDE + lhi * 16;
      v8h lo = *(const v8h*)&Asm[base];       // K = 16g .. 16g+7
      v8h hi = *(const v8h*)&Asm[base + 8];   // K = 16g+8 .. 16g+15
      af[f] = __builtin_shufflevector(lo, hi, 0, 1, 2, 3, 4, 5, 6, 7,
                                      8, 9, 10, 11, 12, 13, 14, 15);
    }

#pragma unroll
    for (int f = 0; f < 2; ++f)
#pragma unroll
      for (int t2 = 0; t2 < 4; ++t2)
        acc[f][t2] = __builtin_amdgcn_wmma_f32_16x16x32_f16(
            false, af[f], false, bf[t2], (short)0, acc[f][t2], false, false);

    if (it < NITER - 1) commitTile(it + 1);
#if HAVE_ASYNC
    wait_async0();
#endif
    __syncthreads();
  }

  // Epilogue: each lane holds 8 consecutive M for one N; NCHW output is
  // contiguous in m except across batch seams.
#pragma unroll
  for (int f = 0; f < 2; ++f) {
    const int mRow = mBase + waveM * 32 + f * 16 + lhi * 8;
    const int bo   = mRow / HWSZ;
    const int hwr  = mRow - bo * HWSZ;
    const bool contig = hwr <= (HWSZ - 8);
#pragma unroll
    for (int t2 = 0; t2 < 4; ++t2) {
      const int o = nBase + t2 * 16 + lrow;
      if (contig) {
        float* dst = out + ((size_t)bo * OUTC + o) * HWSZ + hwr;
        float4 s0 = make_float4(acc[f][t2][0], acc[f][t2][1], acc[f][t2][2], acc[f][t2][3]);
        float4 s1 = make_float4(acc[f][t2][4], acc[f][t2][5], acc[f][t2][6], acc[f][t2][7]);
        ((float4*)dst)[0] = s0;
        ((float4*)dst)[1] = s1;
      } else {
#pragma unroll
        for (int v = 0; v < 8; ++v) {
          const int mm = mRow + v;
          const int b2 = mm / HWSZ;
          const int h2 = mm - b2 * HWSZ;
          out[((size_t)b2 * OUTC + o) * HWSZ + h2] = acc[f][t2][v];
        }
      }
    }
  }
}

// -------------------------------------------------------------------------
extern "C" void kernel_launch(void* const* d_in, const int* in_sizes, int n_in,
                              void* d_out, int out_size, void* d_ws, size_t ws_size,
                              hipStream_t stream) {
  const float*         x   = (const float*)d_in[0];
  const unsigned char* pos = (const unsigned char*)d_in[1];
  const unsigned char* neg = (const unsigned char*)d_in[2];
  float* out = (float*)d_out;

  // Workspace: [ x NHWC f16 : 51,380,224 B ][ packed B fragments : 2,359,296 B ]
  _Float16* xh    = (_Float16*)d_ws;
  _Float16* wfrag = (_Float16*)((char*)d_ws + (size_t)MTOT * CIN * sizeof(_Float16));

  k_nchw_to_nhwc_f16<<<dim3(HWSZ / 64, CIN / 32, BATCH), 256, 0, stream>>>(x, xh);
  k_pack_weight<<<dim3((KTAP * 32 * 8 * 32) / 256), 256, 0, stream>>>(pos, neg, wfrag);
  k_conv_wmma<<<dim3(MTOT / MT, OUTC / NT), 256, 0, stream>>>(xh, wfrag, out);
}
